// MultiHeadAttention_19250043420976
// MI455X (gfx1250) — compile-verified, matrix operand order checked by reading, not run on hardware
//
#include <hip/hip_runtime.h>

// ---------------------------------------------------------------------------
// MI455X (gfx1250) fused multi-head attention, bf16 WMMA + f32 accumulate.
//   K0a: convert x, Wp  -> bf16 (one time)
//   K0b: transpose Wq/Wk/Wv -> bf16 [N, C] (one time, LDS tile transpose)
//   K1 : QKV projection, 32x32 tile/wave -> Q,K [B,H,T,D]; V transposed [B,H,D,T]
//   K2 : causal flash attention, 64-key chunks, exp2-domain online softmax
//   K3 : output projection, 32x32 tile/wave, + bias -> f32 [B,T,C]
// ---------------------------------------------------------------------------

typedef __attribute__((ext_vector_type(16))) __bf16 v16bf;
typedef __attribute__((ext_vector_type(8)))  __bf16 v8bf;
typedef __attribute__((ext_vector_type(8)))  float  v8f;

#define WMMA_BF16(a, b, c) \
  __builtin_amdgcn_wmma_f32_16x16x32_bf16(false, (a), false, (b), (short)0, (c), false, false)

#if __has_builtin(__builtin_amdgcn_exp2f)
#define EXP2F(x) __builtin_amdgcn_exp2f(x)
#else
#define EXP2F(x) __expf((x) * 0.6931471805599453f)
#endif

#define BB 8
#define TT 1024
#define CC 768
#define HH 12
#define DD 64

// A/B fragment (16x32 bf16): lane holds row m = lane%16; K values are two
// contiguous 8-element runs at base0 and base0+16, base0 = (lane<16)?0:8.
static __device__ inline v16bf frag_rows_bf16(const __bf16* rowptr, int base0) {
  v8bf lo = *(const v8bf*)(rowptr + base0);
  v8bf hi = *(const v8bf*)(rowptr + base0 + 16);
  v16bf r;
#pragma unroll
  for (int i = 0; i < 8; ++i) { r[i] = lo[i]; r[i + 8] = hi[i]; }
  return r;
}

// ---------------------------------------------------------------------------
// K0a: linear f32 -> bf16 conversion (8 elements per thread).
// ---------------------------------------------------------------------------
__global__ __launch_bounds__(256) void convert_bf16_kernel(
    const float* __restrict__ src, __bf16* __restrict__ dst, int n8) {
  const int i = blockIdx.x * 256 + threadIdx.x;
  if (i >= n8) return;
  const long base = (long)i * 8;
  v8bf o;
#pragma unroll
  for (int j = 0; j < 8; ++j) o[j] = (__bf16)src[base + j];
  *(v8bf*)(dst + base) = o;
}

// ---------------------------------------------------------------------------
// K0b: Wq/Wk/Wv [C, N] f32 -> [N, C] bf16 via 32x32 LDS tile transpose.
// ---------------------------------------------------------------------------
__global__ __launch_bounds__(256) void transpose_w_kernel(
    const float* __restrict__ Wq, const float* __restrict__ Wk,
    const float* __restrict__ Wv,
    __bf16* __restrict__ oq, __bf16* __restrict__ ok, __bf16* __restrict__ ov) {
  __shared__ float tile[32][33];
  const float* W;
  __bf16* out;
  if (blockIdx.z == 0)      { W = Wq; out = oq; }
  else if (blockIdx.z == 1) { W = Wk; out = ok; }
  else                      { W = Wv; out = ov; }
  const int c0 = blockIdx.x * 32, n0 = blockIdx.y * 32;
  const int tx = threadIdx.x & 31, ty = threadIdx.x >> 5;   // 32x8
#pragma unroll
  for (int j = 0; j < 32; j += 8)
    tile[ty + j][tx] = W[(long)(c0 + ty + j) * CC + n0 + tx];
  __syncthreads();
#pragma unroll
  for (int j = 0; j < 32; j += 8)
    out[(long)(n0 + ty + j) * CC + c0 + tx] = (__bf16)tile[tx][ty + j];
}

// ---------------------------------------------------------------------------
// K1: QKV projection, one 32x32 output tile per wave (2 A frags x 2 B frags
// -> 4 wmma per K-chunk; 2 b128 loads per wmma).
// ---------------------------------------------------------------------------
__global__ __launch_bounds__(256) void qkv_proj_kernel(
    const __bf16* __restrict__ xb,
    const __bf16* __restrict__ wqt, const __bf16* __restrict__ wkt,
    const __bf16* __restrict__ wvt,
    __bf16* __restrict__ qo, __bf16* __restrict__ ko, __bf16* __restrict__ vto) {
  const int lane = threadIdx.x & 31;
  const int wave = threadIdx.x >> 5;
  const int m16 = lane & 15;
  const int half = lane >> 4;
  const int base0 = half ? 8 : 0;

  const int tile = blockIdx.x * 8 + wave;   // (8192/32)*(768/32)=6144 tiles
  const int tn = tile % (CC / 32);
  const int tm = tile / (CC / 32);

  const __bf16* Wt;
  if (blockIdx.y == 0)      Wt = wqt;
  else if (blockIdx.y == 1) Wt = wkt;
  else                      Wt = wvt;

  const int r0 = tm * 32 + m16, r1 = r0 + 16;   // b*T + t
  const int c0 = tn * 32 + m16, c1 = c0 + 16;   // h*D + d
  const __bf16* arow0 = xb + (long)r0 * CC;
  const __bf16* arow1 = xb + (long)r1 * CC;
  const __bf16* brow0 = Wt + (long)c0 * CC;
  const __bf16* brow1 = Wt + (long)c1 * CC;

  v8f a00 = {}, a01 = {}, a10 = {}, a11 = {};
#pragma unroll 2
  for (int kc = 0; kc < CC; kc += 32) {
    const v16bf fa0 = frag_rows_bf16(arow0 + kc, base0);
    const v16bf fa1 = frag_rows_bf16(arow1 + kc, base0);
    const v16bf fb0 = frag_rows_bf16(brow0 + kc, base0);
    const v16bf fb1 = frag_rows_bf16(brow1 + kc, base0);
    a00 = WMMA_BF16(fa0, fb0, a00);
    a01 = WMMA_BF16(fa0, fb1, a01);
    a10 = WMMA_BF16(fa1, fb0, a10);
    a11 = WMMA_BF16(fa1, fb1, a11);
  }

  if (blockIdx.y < 2) {
    __bf16* out = (blockIdx.y == 0) ? qo : ko;   // [B,H,T,D]
    auto store_qk = [&](const v8f& acc, int M16, int col) {
      const int h = col >> 6, d = col & 63;
#pragma unroll
      for (int r = 0; r < 8; ++r) {
        const int M = M16 + r + 8 * half;
        const int b = M >> 10, t = M & (TT - 1);
        out[(((long)(b * HH + h)) * TT + t) * DD + d] = (__bf16)acc[r];
      }
    };
    store_qk(a00, tm * 32,      c0);
    store_qk(a01, tm * 32,      c1);
    store_qk(a10, tm * 32 + 16, c0);
    store_qk(a11, tm * 32 + 16, c1);
  } else {
    // V transposed: vt[b,h,d,t]; 8 consecutive t per lane -> one 16B store.
    auto store_v = [&](const v8f& acc, int M16, int col) {
      const int h = col >> 6, d = col & 63;
      const int M0 = M16 + 8 * half;
      const int b = M0 >> 10, t0 = M0 & (TT - 1);
      v8bf o;
#pragma unroll
      for (int r = 0; r < 8; ++r) o[r] = (__bf16)acc[r];
      *(v8bf*)(vto + (((long)(b * HH + h)) * DD + d) * TT + t0) = o;
    };
    store_v(a00, tm * 32,      c0);
    store_v(a01, tm * 32,      c1);
    store_v(a10, tm * 32 + 16, c0);
    store_v(a11, tm * 32 + 16, c1);
  }
}

// ---------------------------------------------------------------------------
// K2: causal flash attention per (b,h). One wave owns 16 query rows and a
// 16x64 f32 accumulator; 64-key chunks with exp2-domain online softmax.
// nfull unmasked chunks + exactly one masked diagonal chunk per wave.
// ---------------------------------------------------------------------------
__global__ __launch_bounds__(256) void flash_attn_kernel(
    const __bf16* __restrict__ Q, const __bf16* __restrict__ K,
    const __bf16* __restrict__ Vt /* [B,H,D,T] */,
    __bf16* __restrict__ O /* [B,T,C] bf16 */) {
  __shared__ __bf16 Pbuf[8][16 * 64];       // per-wave 2KB transpose buffer

  const int lane = threadIdx.x & 31;
  const int wave = threadIdx.x >> 5;
  const int m16 = lane & 15;
  const int half = lane >> 4;
  const int base0 = half ? 8 : 0;

  const int bh = blockIdx.x >> 3;           // b*H + h
  const int qblk = blockIdx.x & 7;
  const int qbase = qblk * 128 + wave * 16;

  const __bf16* Qb = Q + (long)bh * TT * DD;
  const __bf16* Kb = K + (long)bh * TT * DD;
  const __bf16* Vtb = Vt + (long)bh * DD * TT;

  const __bf16* qrow = Qb + (long)(qbase + m16) * DD;
  const v16bf qa0 = frag_rows_bf16(qrow + 0, base0);
  const v16bf qa1 = frag_rows_bf16(qrow + 32, base0);

  v8f o[4] = {{}, {}, {}, {}};
  float mrow[8], lrow[8];
#pragma unroll
  for (int r = 0; r < 8; ++r) { mrow[r] = -1.0e30f; lrow[r] = 0.f; }

  // Scores kept in log2 domain: ls = D^-0.5 * log2(e).
  const float ls = 0.1803368801111204f;
  __bf16* pb = &Pbuf[wave][0];

  auto process_chunk = [&](int s0, bool masked) {
    // ---- S = Q K^T for four 16-key tiles (8 wmma) ----------------------
    v8f s[4];
#pragma unroll
    for (int t = 0; t < 4; ++t) {
      const __bf16* krow = Kb + (long)(s0 + 16 * t + m16) * DD;
      v8f acc = {};
      acc = WMMA_BF16(qa0, frag_rows_bf16(krow + 0, base0), acc);
      acc = WMMA_BF16(qa1, frag_rows_bf16(krow + 32, base0), acc);
      s[t] = acc;
    }

    // ---- scale (+ causal mask) + row max over 64 keys ------------------
    float p[4][8], mx[8];
#pragma unroll
    for (int r = 0; r < 8; ++r) mx[r] = -1.0e30f;
#pragma unroll
    for (int t = 0; t < 4; ++t) {
#pragma unroll
      for (int r = 0; r < 8; ++r) {
        float v = s[t][r] * ls;
        if (masked) {
          const int qr = qbase + r + 8 * half;
          if (s0 + 16 * t + m16 > qr) v = -1.0e30f;
        }
        p[t][r] = v;
        mx[r] = fmaxf(mx[r], v);
      }
    }
#pragma unroll
    for (int off = 1; off < 16; off <<= 1) {
#pragma unroll
      for (int r = 0; r < 8; ++r) mx[r] = fmaxf(mx[r], __shfl_xor(mx[r], off, 32));
    }

    // ---- online softmax update (exp2 domain) ---------------------------
    float sum[8];
#pragma unroll
    for (int r = 0; r < 8; ++r) {
      const float nm = fmaxf(mrow[r], mx[r]);
      const float cf = EXP2F(mrow[r] - nm);
      mrow[r] = nm;
      float srm = 0.f;
#pragma unroll
      for (int t = 0; t < 4; ++t) {
        const float e = EXP2F(p[t][r] - nm);
        p[t][r] = e;
        srm += e;
      }
      sum[r] = srm;
      lrow[r] *= cf;
#pragma unroll
      for (int f = 0; f < 4; ++f) o[f][r] *= cf;
    }
#pragma unroll
    for (int off = 1; off < 16; off <<= 1) {
#pragma unroll
      for (int r = 0; r < 8; ++r) sum[r] += __shfl_xor(sum[r], off, 32);
    }
#pragma unroll
    for (int r = 0; r < 8; ++r) lrow[r] += sum[r];

    // ---- P (C/D layout) -> LDS -> two A-fragments ----------------------
#pragma unroll
    for (int t = 0; t < 4; ++t) {
#pragma unroll
      for (int r = 0; r < 8; ++r) {
        const int m = r + 8 * half;
        pb[m * 64 + 16 * t + m16] = (__bf16)p[t][r];
      }
    }
    const v16bf pa0 = frag_rows_bf16(pb + m16 * 64, base0);
    const v16bf pa1 = frag_rows_bf16(pb + m16 * 64 + 32, base0);

    // ---- O += P * V : Vt rows contiguous along keys (8 wmma) -----------
#pragma unroll
    for (int f = 0; f < 4; ++f) {
      const __bf16* vrow = Vtb + (long)(f * 16 + m16) * TT + s0;
      o[f] = WMMA_BF16(pa0, frag_rows_bf16(vrow, base0), o[f]);
      o[f] = WMMA_BF16(pa1, frag_rows_bf16(vrow + 32, base0), o[f]);
    }
  };

  // Full (unmasked) 64-key chunks, then exactly one masked diagonal chunk.
  const int nfull = (qbase >= 63) ? (((qbase - 63) >> 6) + 1) : 0;
  for (int i = 0; i < nfull; ++i) {
    const int s0 = i * 64;
    // prefetch next chunk's K rows (global_prefetch_b8)
    __builtin_prefetch(Kb + (long)(s0 + 64 + m16) * DD, 0, 3);
    process_chunk(s0, false);
  }
  process_chunk(nfull * 64, true);

  // ---- normalize and store in concat-head [B,T,C] layout ---------------
  const int b = bh / HH, h = bh % HH;
#pragma unroll
  for (int r = 0; r < 8; ++r) {
    const float inv = 1.0f / lrow[r];
    const int t = qbase + r + 8 * half;
    __bf16* orow = O + ((long)b * TT + t) * CC + h * DD;
#pragma unroll
    for (int f = 0; f < 4; ++f) orow[f * 16 + m16] = (__bf16)(o[f][r] * inv);
  }
}

// ---------------------------------------------------------------------------
// K3: output projection, 32x32 tile per wave. Y = att * Wp^T + bp.
// B(k,n) = Wp[n,k] so B-fragment reads bf16 Wp rows contiguously.
// ---------------------------------------------------------------------------
__global__ __launch_bounds__(256) void out_proj_kernel(
    const __bf16* __restrict__ A /* [B*T, C] bf16 */,
    const __bf16* __restrict__ Wpb /* [C, C] bf16 */,
    const float* __restrict__ bp, float* __restrict__ Y) {
  const int lane = threadIdx.x & 31;
  const int wave = threadIdx.x >> 5;
  const int m16 = lane & 15;
  const int half = lane >> 4;
  const int base0 = half ? 8 : 0;

  const int tile = blockIdx.x * 8 + wave;   // 256*24 tiles of 32x32
  const int tn = tile % (CC / 32);
  const int tm = tile / (CC / 32);

  const int r0 = tm * 32 + m16, r1 = r0 + 16;
  const int c0 = tn * 32 + m16, c1 = c0 + 16;
  const __bf16* arow0 = A + (long)r0 * CC;
  const __bf16* arow1 = A + (long)r1 * CC;
  const __bf16* brow0 = Wpb + (long)c0 * CC;
  const __bf16* brow1 = Wpb + (long)c1 * CC;

  v8f a00 = {}, a01 = {}, a10 = {}, a11 = {};
#pragma unroll 2
  for (int kc = 0; kc < CC; kc += 32) {
    const v16bf fa0 = frag_rows_bf16(arow0 + kc, base0);
    const v16bf fa1 = frag_rows_bf16(arow1 + kc, base0);
    const v16bf fb0 = frag_rows_bf16(brow0 + kc, base0);
    const v16bf fb1 = frag_rows_bf16(brow1 + kc, base0);
    a00 = WMMA_BF16(fa0, fb0, a00);
    a01 = WMMA_BF16(fa0, fb1, a01);
    a10 = WMMA_BF16(fa1, fb0, a10);
    a11 = WMMA_BF16(fa1, fb1, a11);
  }

  const float bias0 = bp[c0];
  const float bias1 = bp[c1];
  auto store_y = [&](const v8f& acc, int M16, int col, float bias) {
#pragma unroll
    for (int r = 0; r < 8; ++r) {
      const int M = M16 + r + 8 * half;
      Y[(long)M * CC + col] = acc[r] + bias;
    }
  };
  store_y(a00, tm * 32,      c0, bias0);
  store_y(a01, tm * 32,      c1, bias1);
  store_y(a10, tm * 32 + 16, c0, bias0);
  store_y(a11, tm * 32 + 16, c1, bias1);
}

// ---------------------------------------------------------------------------
extern "C" void kernel_launch(void* const* d_in, const int* in_sizes, int n_in,
                              void* d_out, int out_size, void* d_ws, size_t ws_size,
                              hipStream_t stream) {
  (void)in_sizes; (void)n_in; (void)out_size; (void)ws_size;
  const float* x  = (const float*)d_in[0];
  const float* Wq = (const float*)d_in[1];
  const float* Wk = (const float*)d_in[2];
  const float* Wv = (const float*)d_in[3];
  const float* Wp = (const float*)d_in[4];
  const float* bp = (const float*)d_in[5];
  float* out = (float*)d_out;

  const long NX   = (long)BB * TT * CC;        // 6291456
  const long NW   = (long)CC * CC;             // 589824
  const long NQKV = (long)BB * HH * TT * DD;   // 6291456

  __bf16* xb  = (__bf16*)d_ws;     // [B*T, C]
  __bf16* wqt = xb + NX;           // [N, C] transposed
  __bf16* wkt = wqt + NW;
  __bf16* wvt = wkt + NW;
  __bf16* wpb = wvt + NW;          // [C, C] straight bf16 copy of Wp
  __bf16* q   = wpb + NW;          // [B,H,T,D]
  __bf16* k   = q + NQKV;
  __bf16* vt  = k + NQKV;          // [B,H,D,T]
  __bf16* att = vt + NQKV;         // [B,T,C]
  // total ≈ 68 MB of workspace

  const int blocks32 = ((BB * TT / 32) * (CC / 32)) / 8;  // 768

  convert_bf16_kernel<<<dim3((int)(NX / 8 / 256)), 256, 0, stream>>>(x, xb, (int)(NX / 8));
  convert_bf16_kernel<<<dim3((int)(NW / 8 / 256)), 256, 0, stream>>>(Wp, wpb, (int)(NW / 8));
  transpose_w_kernel<<<dim3(CC / 32, CC / 32, 3), 256, 0, stream>>>(Wq, Wk, Wv, wqt, wkt, wvt);

  qkv_proj_kernel<<<dim3(blocks32, 3), 256, 0, stream>>>(xb, wqt, wkt, wvt, q, k, vt);
  flash_attn_kernel<<<dim3(BB * HH * (TT / 128)), 256, 0, stream>>>(q, k, vt, att);
  out_proj_kernel<<<dim3(blocks32), 256, 0, stream>>>(att, wpb, bp, out);
}
